// MultiscaleAttention_53068615909784
// MI455X (gfx1250) — matrix operand order repaired and checked by reading, not compiled, not run on hardware
//
#include <hip/hip_runtime.h>
#include <hip/hip_bf16.h>

// ---------------- problem constants ----------------
#define BB 8
#define T0C 8
#define H0C 28
#define W0C 28
#define EE 384
#define DD 384
#define NHD 4
#define HD 96
#define NN 6273          // 1 + 8*28*28
#define NQ 1569          // 1 + 8*14*14
#define NQS 1568         // NQ - 1 (spatial tokens)
#define M_QKV (BB*NN)    // 50184
#define M_PROJ (BB*NQ)   // 12552
#define QT16 99          // ceil(NQ/16)
#define KT16 100         // padded key tiles of 16
#define NK_PAD 1600      // padded key count (LDS / Vt row stride)
#define ATS 392          // LDS A-tile row stride in halfs (384 + 8 pad)
#define SCALE_QK 0.10206207261596575f  // 1/sqrt(96)

typedef __attribute__((ext_vector_type(16))) _Float16 v16h;
typedef __attribute__((ext_vector_type(8)))  _Float16 v8h;
typedef __attribute__((ext_vector_type(4)))  _Float16 v4h;
typedef __attribute__((ext_vector_type(8)))  float    v8f;
typedef __attribute__((ext_vector_type(4)))  int      v4i_t;

#if defined(__has_builtin)
#if __has_builtin(__builtin_amdgcn_global_load_async_to_lds_b128)
#define HAS_ASYNC 1
#endif
#endif
#ifndef HAS_ASYNC
#define HAS_ASYNC 0
#endif

// async global->LDS 16-byte copy (ASYNCcnt path); fallback: regular copy
__device__ inline void async_cp16(const _Float16* g, _Float16* l) {
#if HAS_ASYNC
  __builtin_amdgcn_global_load_async_to_lds_b128(
      (__attribute__((address_space(1))) v4i_t*)g,
      (__attribute__((address_space(3))) v4i_t*)l, 0, 0);
#else
  *(v8h*)l = *(const v8h*)g;
#endif
}
__device__ inline void async_join() {
#if HAS_ASYNC
  asm volatile("s_wait_asynccnt 0" ::: "memory");
#endif
}

__device__ inline v8f zero8() {
  v8f z;
#pragma unroll
  for (int i = 0; i < 8; ++i) z[i] = 0.f;
  return z;
}

__device__ inline v8f wmma_f16(v16h a, v16h b, v8f c) {
  // D = A(16x32) * B(32x16) + C, f32 accumulate
  return __builtin_amdgcn_wmma_f32_16x16x32_f16(
      false, a, false, b, (short)0, c, false, false);
}

// A fragment 16x32 from row-major f16 (ld halfs per row), rows clamped.
// lane (m=lane&15, h2=lane>>4): elems 0-7 -> K=k0+h2*8+i, elems 8-15 -> +16
__device__ inline v16h ldA_f16(const _Float16* __restrict__ base, int ld,
                               int row0, int k0, int rowClamp) {
  int lane = threadIdx.x & 31;
  int r = row0 + (lane & 15);
  r = (r < rowClamp) ? r : (rowClamp - 1);
  int h2 = lane >> 4;
  const _Float16* p = base + (size_t)r * ld + k0 + h2 * 8;
  v8h lo = *(const v8h*)(p);        // 16B
  v8h hi = *(const v8h*)(p + 16);   // 16B
  v16h f;
#pragma unroll
  for (int i = 0; i < 8; ++i) { f[i] = lo[i]; f[8 + i] = hi[i]; }
  return f;
}

// B fragment 32x16 from "N x K" row-major f16: math-B[k][n] = src[n][k]
// lane (n=lane&15, h2=lane>>4): elems 0-7 -> K=k0+h2*8+i, elems 8-15 -> +16
__device__ inline v16h ldB_f16(const _Float16* __restrict__ base, int ld,
                               int n0, int k0, int nClamp) {
  int lane = threadIdx.x & 31;
  int n = n0 + (lane & 15);
  n = (n < nClamp) ? n : (nClamp - 1);
  int h2 = lane >> 4;
  const _Float16* p = base + (size_t)n * ld + k0 + h2 * 8;
  v8h lo = *(const v8h*)(p);
  v8h hi = *(const v8h*)(p + 16);
  v16h f;
#pragma unroll
  for (int i = 0; i < 8; ++i) { f[i] = lo[i]; f[8 + i] = hi[i]; }
  return f;
}

// ---------------- kernel 0: f32 -> f16 convert (vectorized x4) ----------------
__global__ void cvt_kernel(const float* __restrict__ in,
                           _Float16* __restrict__ out, int n4) {
  int i = blockIdx.x * blockDim.x + threadIdx.x;
  if (i >= n4) return;
  float4 v = ((const float4*)in)[i];
  v4h o;
  o[0] = (_Float16)v.x; o[1] = (_Float16)v.y;
  o[2] = (_Float16)v.z; o[3] = (_Float16)v.w;
  ((v4h*)out)[i] = o;
}

// ---------------- kernel 1: QKV projection GEMM (all f16, LDS-staged A) ----------------
// Y(f16)[M_QKV][1152] = X[M_QKV][384] * qkv_w^T + qkv_b ; 2 M-tiles per wave
__global__ __launch_bounds__(256) void qkv_gemm(
    const _Float16* __restrict__ X, const _Float16* __restrict__ W,
    const float* __restrict__ bias, _Float16* __restrict__ Y) {
  __shared__ __align__(16) _Float16 At[32 * ATS];
  int m0 = blockIdx.y * 32;
  // stage 32 x 384 A tile via async global->LDS copies (6 per thread)
  for (int c = threadIdx.x; c < 32 * 48; c += 256) {
    int row = c / 48, c8 = (c % 48) * 8;
    int r = m0 + row;
    r = r < M_QKV ? r : M_QKV - 1;
    async_cp16(X + (size_t)r * EE + c8, &At[row * ATS + c8]);
  }
  async_join();
  __syncthreads();
  int wave = threadIdx.x >> 5;
  int nt = blockIdx.x * 8 + wave;       // 72 n-tiles total, grid.x = 9
  int n0 = nt * 16;
  v8f acc0 = zero8(), acc1 = zero8();
#pragma unroll
  for (int k0 = 0; k0 < EE; k0 += 32) {
    v16h b = ldB_f16(W, EE, n0, k0, 3 * DD);
    v16h a0 = ldA_f16(At, ATS, 0, k0, 32);   // ds_load_b128 pairs
    v16h a1 = ldA_f16(At, ATS, 16, k0, 32);
    acc0 = wmma_f16(a0, b, acc0);
    acc1 = wmma_f16(a1, b, acc1);
  }
  int lane = threadIdx.x & 31, sub = lane & 15, h2 = lane >> 4;
  int cc = n0 + sub;
  float bv = bias[cc];
#pragma unroll
  for (int j = 0; j < 8; ++j) {
    int r0 = m0 + h2 * 8 + j;
    int r1 = r0 + 16;
    if (r0 < M_QKV) Y[(size_t)r0 * 1152 + cc] = (_Float16)(acc0[j] + bv);
    if (r1 < M_QKV) Y[(size_t)r1 * 1152 + cc] = (_Float16)(acc1[j] + bv);
  }
}

// ---------------- kernel 2: depthwise 3x3x3 pool (stride 1,2,2) + LayerNorm ----------------
__global__ void pool_ln_kernel(const _Float16* __restrict__ qkv,
                               const float* __restrict__ w,
                               const float* __restrict__ g,
                               const float* __restrict__ beta,
                               float* __restrict__ outf,
                               _Float16* __restrict__ outh,
                               _Float16* __restrict__ vt, int s) {
  __shared__ float sh[HD];
  __shared__ float stats[2];
  int bid = blockIdx.x;
  int bh = bid / NQ;
  int tok = bid % NQ;
  int b = bh / NHD, h = bh % NHD;
  int c = threadIdx.x;
  size_t colbase = (size_t)s * 384 + (size_t)h * 96;
  float val;
  if (tok == 0) {
    val = (float)qkv[((size_t)b * NN) * 1152 + colbase + c];
  } else {
    int q3 = tok - 1;
    int t = q3 / 196, yy = (q3 % 196) / 14, xx = q3 % 14;
    float sum = 0.f;
#pragma unroll
    for (int dt = 0; dt < 3; ++dt) {
      int ti = t + dt - 1;
      if (ti < 0 || ti >= T0C) continue;
#pragma unroll
      for (int dy = 0; dy < 3; ++dy) {
        int yi = 2 * yy + dy - 1;
        if (yi < 0 || yi >= H0C) continue;
#pragma unroll
        for (int dx = 0; dx < 3; ++dx) {
          int xi = 2 * xx + dx - 1;
          if (xi < 0 || xi >= W0C) continue;
          int n = 1 + (ti * H0C + yi) * W0C + xi;
          sum += (float)qkv[((size_t)b * NN + n) * 1152 + colbase + c] *
                 w[c * 27 + dt * 9 + dy * 3 + dx];
        }
      }
    }
    val = sum;
  }
  sh[c] = val;
  __syncthreads();
  if (c == 0) {
    float m = 0.f;
    for (int i = 0; i < HD; ++i) m += sh[i];
    m /= HD;
    float v = 0.f;
    for (int i = 0; i < HD; ++i) { float d = sh[i] - m; v += d * d; }
    v /= HD;
    stats[0] = m; stats[1] = v;
  }
  __syncthreads();
  float m = stats[0], v = stats[1];
  float normed = (val - m) * rsqrtf(v + 1e-5f) * g[c] + beta[c];
  if (outf) outf[((size_t)bh * NQ + tok) * HD + c] = normed;
  if (outh) outh[((size_t)bh * NQ + tok) * HD + c] = (_Float16)normed;
  if (vt)   vt[((size_t)bh * HD + c) * NK_PAD + tok] = (_Float16)normed;
}

// ---------------- kernel 3: relative-position dot products ----------------
__global__ void rel_kernel(const float* __restrict__ qp,
                           const float* __restrict__ rh,
                           const float* __restrict__ rw,
                           const float* __restrict__ rt,
                           float* __restrict__ oh, float* __restrict__ ow,
                           float* __restrict__ ot) {
  int bid = blockIdx.x;
  int bh = bid / NQS;
  int q = bid % NQS;
  int qt = q / 196, qh = (q % 196) / 14, qw = q % 14;
  const float* qv = qp + ((size_t)bh * NQ + 1 + q) * HD;
  int t = threadIdx.x;
  if (t < 14) {
    const float* r = rh + (size_t)(qh - t + 13) * HD;
    float s = 0.f;
    for (int c = 0; c < HD; ++c) s += qv[c] * r[c];
    oh[((size_t)bh * NQS + q) * 14 + t] = s;
  } else if (t < 28) {
    int kw = t - 14;
    const float* r = rw + (size_t)(qw - kw + 13) * HD;
    float s = 0.f;
    for (int c = 0; c < HD; ++c) s += qv[c] * r[c];
    ow[((size_t)bh * NQS + q) * 14 + kw] = s;
  } else if (t < 36) {
    int kt = t - 28;
    const float* r = rt + (size_t)(qt - kt + 7) * HD;
    float s = 0.f;
    for (int c = 0; c < HD; ++c) s += qv[c] * r[c];
    ot[((size_t)bh * NQS + q) * 8 + kt] = s;
  }
}

// ---------------- kernel 4: fused attention ----------------
// one block per (b, h, 16-row q tile).
// LDS: S f32[16][1600] | sums f32[16] | relS f32[16][36] | P f16[16][1600]
#define LDS_SUMS (16 * NK_PAD)
#define LDS_RELS (LDS_SUMS + 16)
#define LDS_PH_F (LDS_RELS + 16 * 36)           // float offset where P panel starts
#define LDS_BYTES ((size_t)LDS_PH_F * 4 + (size_t)16 * NK_PAD * 2)

__global__ __launch_bounds__(256) void attn_kernel(
    const float* __restrict__ qp, const _Float16* __restrict__ qh,
    const _Float16* __restrict__ kh16, const _Float16* __restrict__ vt,
    const float* __restrict__ relh, const float* __restrict__ relw,
    const float* __restrict__ relt, _Float16* __restrict__ O) {
  extern __shared__ float smem[];
  float* S = smem;
  float* sums = smem + LDS_SUMS;
  float* relS = smem + LDS_RELS;
  _Float16* Ph = (_Float16*)(smem + LDS_PH_F);
  int bh = blockIdx.x / QT16;
  int qtile = blockIdx.x % QT16;
  int q0 = qtile * 16;
  const float* qb = qp + (size_t)bh * NQ * HD;
  const _Float16* qhb = qh + (size_t)bh * NQ * HD;
  const _Float16* khb = kh16 + (size_t)bh * NQ * HD;
  const _Float16* vtb = vt + (size_t)bh * HD * NK_PAD;
  const float* rhb = relh + (size_t)bh * NQS * 14;
  const float* rwb = relw + (size_t)bh * NQS * 14;
  const float* rtb = relt + (size_t)bh * NQS * 8;
  int wave = threadIdx.x >> 5;
  int lane = threadIdx.x & 31;
  int sub = lane & 15;
  int h2 = lane >> 4;

  // ---- stage rel-pos rows for this q tile into LDS (16 rows x 36 vals) ----
  for (int idx = threadIdx.x; idx < 16 * 36; idx += 256) {
    int row = idx / 36, t = idx % 36;
    int qq = q0 + row - 1;
    qq = qq < 0 ? 0 : (qq < NQS ? qq : NQS - 1);
    float v;
    if (t < 14)      v = rhb[(size_t)qq * 14 + t];
    else if (t < 28) v = rwb[(size_t)qq * 14 + (t - 14)];
    else             v = rtb[(size_t)qq * 8 + (t - 28)];
    relS[idx] = v;
  }

  // ---- phase A: S = scale * q @ k^T + relpos (or -1e30 pad) ----
  v16h aq0 = ldA_f16(qhb, HD, q0, 0, NQ);
  v16h aq1 = ldA_f16(qhb, HD, q0, 32, NQ);
  v16h aq2 = ldA_f16(qhb, HD, q0, 64, NQ);
  __syncthreads();   // relS ready
  for (int kt16 = wave; kt16 < KT16; kt16 += 8) {
    int kbase = kt16 * 16;
    int ck = kbase + sub;  // this lane's key column
    v8f acc = zero8();
    acc = wmma_f16(aq0, ldB_f16(khb, HD, kbase, 0, NQ), acc);
    acc = wmma_f16(aq1, ldB_f16(khb, HD, kbase, 32, NQ), acc);
    acc = wmma_f16(aq2, ldB_f16(khb, HD, kbase, 64, NQ), acc);
    int kk = ck - 1;
    kk = kk < 0 ? 0 : (kk < NQS ? kk : NQS - 1);
    int kt2 = kk / 196, krem = kk % 196;
    int khh = krem / 14, kww = krem % 14;
    bool ckrel = (ck >= 1) && (ck < NQ);
    bool ckval = (ck < NQ);
#pragma unroll
    for (int j = 0; j < 8; ++j) {
      int row = h2 * 8 + j;
      int gq = q0 + row;
      float relv = relS[row * 36 + khh] + relS[row * 36 + 14 + kww] +
                   relS[row * 36 + 28 + kt2];
      bool userel = ckrel && (gq >= 1) && (gq < NQ);
      float val = acc[j] * SCALE_QK + (userel ? relv : 0.f);
      S[row * NK_PAD + ck] = ckval ? val : -1e30f;
    }
  }
  __syncthreads();

  // ---- phase B: rowwise softmax; write exp() as f16 panel, keep row sums ----
  {
    int row = wave * 2 + h2;          // 8 waves x 2 rows
    float* Sr = S + row * NK_PAD;
    _Float16* Pr = Ph + row * NK_PAD;
    float m = -1e30f;
    for (int i = 0; i < KT16; ++i) m = fmaxf(m, Sr[sub + i * 16]);
#pragma unroll
    for (int off = 8; off >= 1; off >>= 1) m = fmaxf(m, __shfl_xor(m, off, 16));
    float s = 0.f;
    for (int i = 0; i < KT16; ++i) {
      float e = __expf(Sr[sub + i * 16] - m);
      Pr[sub + i * 16] = (_Float16)e;
      s += e;
    }
#pragma unroll
    for (int off = 8; off >= 1; off >>= 1) s += __shfl_xor(s, off, 16);
    if (sub == 0) sums[row] = s;
  }
  __syncthreads();

  // ---- phase C: O = (P @ V) / rowsum + residual q ----
  if (wave < 6) {                     // 6 n-tiles cover HD = 96
    int hd0 = wave * 16;
    v8f acc = zero8();
    for (int ks = 0; ks < NK_PAD / 32; ++ks) {
      v16h a = ldA_f16(Ph, NK_PAD, 0, ks * 32, 16);      // LDS f16 b128 reads
      v16h b = ldB_f16(vtb, NK_PAD, hd0, ks * 32, HD);   // contiguous Vt reads
      acc = wmma_f16(a, b, acc);
    }
    int hd = hd0 + sub;
#pragma unroll
    for (int j = 0; j < 8; ++j) {
      int row = h2 * 8 + j;
      int gq = q0 + row;
      int gqc = gq < NQ ? gq : NQ - 1;
      float res = qb[(size_t)gqc * HD + hd];             // unconditional, clamped
      float val = acc[j] * (1.0f / sums[row]) + ((gq >= 1) ? res : 0.f);
      if (gq < NQ) O[((size_t)bh * NQ + gq) * HD + hd] = (_Float16)val;
    }
  }
}

// ---------------- kernel 5: output projection GEMM (LDS-staged A) ----------------
// out[b*NQ+q][n] = sum_k O[(b, h=k/96, q, hd=k%96)] * proj_w[n][k] + proj_b[n]
__global__ __launch_bounds__(256) void proj_gemm(
    const _Float16* __restrict__ O, const _Float16* __restrict__ W,
    const float* __restrict__ bias, float* __restrict__ out) {
  __shared__ __align__(16) _Float16 At[32 * ATS];
  int m0 = blockIdx.y * 32;
  // stage 32 x 384 A tile; head-transpose gather folded into staging
  for (int c = threadIdx.x; c < 32 * 48; c += 256) {
    int row = c / 48, c8 = (c % 48) * 8;   // ka = c8, 8-half run inside one head
    int r = m0 + row;
    r = r < M_PROJ ? r : M_PROJ - 1;
    int b2 = r / NQ, qi = r % NQ;
    int h = c8 / HD, hd = c8 % HD;
    const _Float16* src =
        O + ((size_t)(b2 * NHD + h) * NQ + qi) * HD + hd;
    async_cp16(src, &At[row * ATS + c8]);
  }
  async_join();
  __syncthreads();
  int wave = threadIdx.x >> 5;
  int nt = blockIdx.x * 8 + wave;     // 24 n-tiles, grid.x = 3
  int n0 = nt * 16;
  v8f acc0 = zero8(), acc1 = zero8();
#pragma unroll
  for (int k0 = 0; k0 < DD; k0 += 32) {
    v16h b = ldB_f16(W, DD, n0, k0, DD);
    v16h a0 = ldA_f16(At, ATS, 0, k0, 32);
    v16h a1 = ldA_f16(At, ATS, 16, k0, 32);
    acc0 = wmma_f16(a0, b, acc0);
    acc1 = wmma_f16(a1, b, acc1);
  }
  int lane = threadIdx.x & 31, sub = lane & 15, h2 = lane >> 4;
  int cc = n0 + sub;
  float bv = bias[cc];
#pragma unroll
  for (int j = 0; j < 8; ++j) {
    int r0 = m0 + h2 * 8 + j;
    int r1 = r0 + 16;
    if (r0 < M_PROJ) out[(size_t)r0 * DD + cc] = acc0[j] + bv;
    if (r1 < M_PROJ) out[(size_t)r1 * DD + cc] = acc1[j] + bv;
  }
}

// ---------------- host launcher ----------------
extern "C" void kernel_launch(void* const* d_in, const int* in_sizes, int n_in,
                              void* d_out, int out_size, void* d_ws,
                              size_t ws_size, hipStream_t stream) {
  (void)in_sizes; (void)n_in; (void)out_size; (void)ws_size;
  const float* x      = (const float*)d_in[0];
  const float* qkv_w  = (const float*)d_in[1];
  const float* qkv_b  = (const float*)d_in[2];
  const float* proj_w = (const float*)d_in[3];
  const float* proj_b = (const float*)d_in[4];
  const float* convq  = (const float*)d_in[5];
  const float* convk  = (const float*)d_in[6];
  const float* convv  = (const float*)d_in[7];
  const float* lnq_g  = (const float*)d_in[8];
  const float* lnq_b  = (const float*)d_in[9];
  const float* lnk_g  = (const float*)d_in[10];
  const float* lnk_b  = (const float*)d_in[11];
  const float* lnv_g  = (const float*)d_in[12];
  const float* lnv_b  = (const float*)d_in[13];
  const float* rel_h  = (const float*)d_in[14];
  const float* rel_w  = (const float*)d_in[15];
  const float* rel_t  = (const float*)d_in[16];

  char* ws = (char*)d_ws;
  size_t off = 0;
  auto alloc = [&](size_t bytes) {
    char* p = ws + off;
    off = (off + bytes + 255) & ~(size_t)255;
    return p;
  };
  _Float16* qkv_h = (_Float16*)alloc((size_t)M_QKV * 1152 * 2);
  _Float16* xh    = (_Float16*)alloc((size_t)M_QKV * EE * 2);
  _Float16* wqkvh = (_Float16*)alloc((size_t)3 * DD * EE * 2);
  _Float16* wprjh = (_Float16*)alloc((size_t)DD * DD * 2);
  float*    qpf   = (float*)   alloc((size_t)BB * NHD * NQ * HD * 4);
  _Float16* qph   = (_Float16*)alloc((size_t)BB * NHD * NQ * HD * 2);
  _Float16* kph   = (_Float16*)alloc((size_t)BB * NHD * NQ * HD * 2);
  _Float16* vth   = (_Float16*)alloc((size_t)BB * NHD * HD * NK_PAD * 2);
  float*    relhb = (float*)   alloc((size_t)BB * NHD * NQS * 14 * 4);
  float*    relwb = (float*)   alloc((size_t)BB * NHD * NQS * 14 * 4);
  float*    reltb = (float*)   alloc((size_t)BB * NHD * NQS * 8 * 4);
  _Float16* Obuf  = (_Float16*)alloc((size_t)BB * NHD * NQ * HD * 2);

  // pre-convert activations + weights to f16
  {
    int n4 = M_QKV * EE / 4;
    cvt_kernel<<<(n4 + 255) / 256, 256, 0, stream>>>(x, xh, n4);
    n4 = 3 * DD * EE / 4;
    cvt_kernel<<<(n4 + 255) / 256, 256, 0, stream>>>(qkv_w, wqkvh, n4);
    n4 = DD * DD / 4;
    cvt_kernel<<<(n4 + 255) / 256, 256, 0, stream>>>(proj_w, wprjh, n4);
  }

  qkv_gemm<<<dim3(9, (M_QKV + 31) / 32), 256, 0, stream>>>(xh, wqkvh, qkv_b, qkv_h);
  pool_ln_kernel<<<BB * NHD * NQ, HD, 0, stream>>>(qkv_h, convq, lnq_g, lnq_b,
                                                   qpf, qph, nullptr, 0);
  pool_ln_kernel<<<BB * NHD * NQ, HD, 0, stream>>>(qkv_h, convk, lnk_g, lnk_b,
                                                   nullptr, kph, nullptr, 1);
  pool_ln_kernel<<<BB * NHD * NQ, HD, 0, stream>>>(qkv_h, convv, lnv_g, lnv_b,
                                                   nullptr, nullptr, vth, 2);
  rel_kernel<<<BB * NHD * NQS, 64, 0, stream>>>(qpf, rel_h, rel_w, rel_t,
                                                relhb, relwb, reltb);
  attn_kernel<<<BB * NHD * QT16, 256, LDS_BYTES, stream>>>(
      qpf, qph, kph, vth, relhb, relwb, reltb, Obuf);
  proj_gemm<<<dim3(3, (M_PROJ + 31) / 32), 256, 0, stream>>>(
      Obuf, wprjh, proj_b, (float*)d_out);
}
